// CodeBook_71021579207117
// MI455X (gfx1250) — matrix operand order repaired and checked by reading, not compiled, not run on hardware
//
#include <hip/hip_runtime.h>
#include <hip/hip_bf16.h>
#include <math.h>

// ---------------- problem constants ----------------
#define NSAMP   32768      // 4*8*32*32
#define DIM     128
#define NCLUST  10000
#define NPAD    10048      // pad to multiple of STAGE_N (64)
#define EMA     0.99f
#define EPSV    1e-05f
#define BETAV   0.25f

// ---- K2 staging geometry ----
#define STAGE_NT 4                  // 16-cluster n-tiles per stage
#define STAGE_N  (STAGE_NT * 16)    // 64 clusters per stage
#define NSTAGES  (NPAD / STAGE_N)   // 157
#define SROW     (DIM + 8)          // LDS row stride in halves: +4-dword skew -> no bank conflicts

typedef __attribute__((ext_vector_type(16))) _Float16 v16h;
typedef __attribute__((ext_vector_type(8)))  float    v8f;
typedef __attribute__((ext_vector_type(4)))  float    v4f;

#if defined(__gfx1250__) && __has_builtin(__builtin_amdgcn_global_load_async_to_lds_b128)
#define USE_ASYNC_COPY 1
#else
#define USE_ASYNC_COPY 0
#endif

#if USE_ASYNC_COPY
// Builtin prototype (from hipcc diagnostic): param1 is a non-const
// "__attribute__((__vector_size__(16))) int __device__ *" (global/AS1 int4*).
typedef int i32x4 __attribute__((vector_size(16)));
typedef __attribute__((address_space(1))) i32x4 as1_i32x4;
typedef __attribute__((address_space(3))) i32x4 as3_i32x4;

// Generic -> AS1: generic address of a global object == global address.
__device__ __forceinline__ as1_i32x4* to_global_v4(const void* p) {
    return (as1_i32x4*)(unsigned long long)(size_t)p;
}
// Generic -> AS3: low 32 bits of the flat address are the LDS offset
// AS3 pointers are 32-bit.
__device__ __forceinline__ as3_i32x4* to_lds_v4(void* p) {
    return (as3_i32x4*)(unsigned int)(size_t)p;
}
#endif

// ---------------- workspace layout (bytes, 64B aligned) ----------------
#define OFF_EF16    0u                       // NPAD*DIM*2   = 2,572,288
#define OFF_EN2     2572288u                 // NPAD*4       =    40,192
#define OFF_IDX     2612480u                 // NSAMP*4      =   131,072
#define OFF_COUNTS  2743552u                 // NCLUST*4     =    40,000
#define OFF_SUMS    2783552u                 // NCLUST*DIM*4 = 5,120,000
#define OFF_NSIZE   7903552u                 // NCLUST*4     =    40,000
#define OFF_NMEAN   7943552u                 // NCLUST*DIM*4 = 5,120,000
#define OFF_SCAL    13063552u                // 2 floats: n_sample, loss

// d_out layout: [ outputs NSAMP*DIM f32 | idx NSAMP (as f32) | loss 1 ]
#define OUT_IDX_OFF  (NSAMP * DIM)
#define OUT_LOSS_OFF (NSAMP * DIM + NSAMP)

// ============ K1: codebook f32 -> f16, ||e||^2 (pad rows: 0 / +inf) ============
__global__ __launch_bounds__(128) void k_prep_clusters(
    const float* __restrict__ cm, _Float16* __restrict__ ef16, float* __restrict__ en2)
{
    const int k = blockIdx.x;          // 0..NPAD-1
    const int d = threadIdx.x;         // 0..127
    float v = 0.0f;
    if (k < NCLUST) v = cm[(size_t)k * DIM + d];
    ef16[(size_t)k * DIM + d] = (_Float16)v;
    float sq = v * v;
    #pragma unroll
    for (int off = 16; off >= 1; off >>= 1) sq += __shfl_xor(sq, off, 32);
    __shared__ float red[4];
    if ((threadIdx.x & 31) == 0) red[threadIdx.x >> 5] = sq;
    __syncthreads();
    if (threadIdx.x == 0) {
        float s = red[0] + red[1] + red[2] + red[3];
        en2[k] = (k < NCLUST) ? s : __builtin_inff();
    }
}

// ---- async-copy helpers ----
#if USE_ASYNC_COPY
template <int N>
__device__ __forceinline__ void wait_async()
{
#if __has_builtin(__builtin_amdgcn_s_wait_asynccnt)
    __builtin_amdgcn_s_wait_asynccnt((unsigned short)N);
#else
    asm volatile("s_wait_asynccnt %0" :: "n"(N) : "memory");
#endif
}
#else
template <int N>
__device__ __forceinline__ void wait_async() {}
#endif

// Stage copy: 64 clusters x 128 halves (16KB) -> LDS with +8-half row skew.
// 256 threads: 4 threads/row, 32 halves (= 4 x b128) per thread, fully coalesced.
__device__ __forceinline__ void copy_stage(const _Float16* __restrict__ ef16,
                                           _Float16* lbuf, int s, int tid)
{
    const int r = tid >> 2;   // stage-local row 0..63
    const int p = tid & 3;    // 32-half part of the row
    const _Float16* g = ef16 + ((size_t)(s * STAGE_N + r) * DIM + p * 32);
    _Float16* l = lbuf + (r * SROW + p * 32);
#if USE_ASYNC_COPY
    #pragma unroll
    for (int q = 0; q < 4; ++q)
        __builtin_amdgcn_global_load_async_to_lds_b128(
            to_global_v4(g + q * 8), to_lds_v4(l + q * 8), 0, 0);
#else
    #pragma unroll
    for (int q = 0; q < 4; ++q)
        *(uint4*)(l + q * 8) = *(const uint4*)(g + q * 8);
#endif
}

// ============ K2: fused f16-WMMA distance GEMM + argmin over K ============
// Block: 8 waves x 16-row A tiles (registers). Codebook streamed through LDS,
// double-buffered via gfx1250 async global->LDS; score = ||e||^2 - 2 x.e.
__global__ __launch_bounds__(256) void k_argmin_gemm(
    const float* __restrict__ samples,        // [NSAMP][DIM] f32
    const _Float16* __restrict__ ef16,        // [NPAD][DIM] f16
    const float* __restrict__ en2,            // [NPAD]
    int* __restrict__ idx_out)                // [NSAMP]
{
    __shared__ _Float16 bsh[2][STAGE_N * SROW];   // 2 x 17,408 B

    const int tid  = threadIdx.x;
    const int lane = tid & 31;
    const int wave = tid >> 5;                          // 8 waves/block
    const int m0   = (blockIdx.x * 8 + wave) * 16;      // 256 blocks -> 32768 rows
    const int half = (lane >> 4) & 1;                   // 0: lanes 0-15, 1: 16-31
    const int l16  = lane & 15;

    // ---- A tile (16x128) -> registers in ISA A-layout, f32 -> f16 inline ----
    const float* ap = samples + (size_t)(m0 + l16) * DIM;
    v16h a[4];
    #pragma unroll
    for (int c = 0; c < 4; ++c) {
        const int o1 = c * 32 + half * 8;
        const int o2 = c * 32 + 16 + half * 8;
        v4f x0 = *(const v4f*)(ap + o1);
        v4f x1 = *(const v4f*)(ap + o1 + 4);
        v4f x2 = *(const v4f*)(ap + o2);
        v4f x3 = *(const v4f*)(ap + o2 + 4);
        #pragma unroll
        for (int j = 0; j < 4; ++j) {
            a[c][j]      = (_Float16)x0[j];
            a[c][4 + j]  = (_Float16)x1[j];
            a[c][8 + j]  = (_Float16)x2[j];
            a[c][12 + j] = (_Float16)x3[j];
        }
    }

    float minv[8];
    int   minn[8];
    #pragma unroll
    for (int r = 0; r < 8; ++r) { minv[r] = __builtin_inff(); minn[r] = 0; }

    copy_stage(ef16, &bsh[0][0], 0, tid);

    for (int s = 0; s < NSTAGES; ++s) {
        const _Float16* cur = &bsh[s & 1][0];
        if (s + 1 < NSTAGES) {             // overlap next copy with this compute
            copy_stage(ef16, &bsh[(s + 1) & 1][0], s + 1, tid);
            wait_async<4>();               // stage-s loads done; next 4 in flight
        } else {
            wait_async<0>();
        }
        __syncthreads();                   // stage-s data visible block-wide

        #pragma unroll
        for (int tl = 0; tl < STAGE_NT; ++tl) {
            const int bcol = s * STAGE_N + tl * 16 + l16;   // this lane's cluster
            const _Float16* bp = cur + (tl * 16 + l16) * SROW;

            // Load ALL four K-chunks into distinct registers first so the
            // ds_load_b128s issue back-to-back and the WMMA chain only needs
            // partial dscnt waits (pipelined LDS latency).
            union bbt { uint4 u4[2]; v16h h; };
            bbt bb[4];
            #pragma unroll
            for (int c = 0; c < 4; ++c) {
                const _Float16* lp = bp + c * 32 + half * 8;
                bb[c].u4[0] = *(const uint4*)(lp);
                bb[c].u4[1] = *(const uint4*)(lp + 16);
            }

            v8f acc = {0.f,0.f,0.f,0.f,0.f,0.f,0.f,0.f};
            #pragma unroll
            for (int c = 0; c < 4; ++c)
                acc = __builtin_amdgcn_wmma_f32_16x16x32_f16(
                    false, a[c], false, bb[c].h, (short)0, acc, false, false);

            const float e2 = en2[bcol];
            // C layout: acc[r] = dot(row m0 + r + 8*half, cluster bcol)
            #pragma unroll
            for (int r = 0; r < 8; ++r) {
                const float sc = fmaf(-2.0f, acc[r], e2);
                if (sc < minv[r]) { minv[r] = sc; minn[r] = bcol; }
            }
        }
        __syncthreads();                   // done reading cur before it is reused
    }

    // ---- cross-lane argmin within each 16-lane half (wave32 shfl_xor) ----
    #pragma unroll
    for (int r = 0; r < 8; ++r) {
        float v = minv[r];
        int   n = minn[r];
        #pragma unroll
        for (int off = 8; off >= 1; off >>= 1) {
            const float ov = __shfl_xor(v, off, 32);
            const int   on = __shfl_xor(n, off, 32);
            if (ov < v || (ov == v && on < n)) { v = ov; n = on; }
        }
        if (l16 == 0) idx_out[m0 + r + 8 * half] = n;
    }
}

// ============ K3: zero accumulators (counts, sums, scalars) ============
__global__ __launch_bounds__(256) void k_zero(
    float* __restrict__ counts, float* __restrict__ sums, float* __restrict__ scal)
{
    const int t = blockIdx.x * 256 + threadIdx.x;
    const int stride = gridDim.x * 256;
    if (t < NCLUST) counts[t] = 0.0f;
    for (int i = t; i < NCLUST * DIM; i += stride) sums[i] = 0.0f;
    if (t < 2) scal[t] = 0.0f;
}

// ============ K4: segment-sum scatter via f32 atomics ============
__global__ __launch_bounds__(128) void k_scatter(
    const float* __restrict__ samples, const int* __restrict__ idx,
    float* __restrict__ counts, float* __restrict__ sums)
{
    const int i = blockIdx.x;      // sample
    const int d = threadIdx.x;     // dim
    const int k = idx[i];
    atomicAdd(&sums[(size_t)k * DIM + d], samples[(size_t)i * DIM + d]);
    if (d == 0) atomicAdd(&counts[k], 1.0f);
}

// ============ K5: EMA size + n_sample reduction ============
__global__ __launch_bounds__(256) void k_ema_size(
    const float* __restrict__ cluster_size, const float* __restrict__ counts,
    float* __restrict__ new_size, float* __restrict__ scal)
{
    const int k = blockIdx.x * 256 + threadIdx.x;
    float ns = 0.0f;
    if (k < NCLUST) {
        ns = cluster_size[k] * EMA + counts[k] * (1.0f - EMA);
        new_size[k] = ns;
    }
    #pragma unroll
    for (int off = 16; off >= 1; off >>= 1) ns += __shfl_xor(ns, off, 32);
    __shared__ float red[8];
    if ((threadIdx.x & 31) == 0) red[threadIdx.x >> 5] = ns;
    __syncthreads();
    if (threadIdx.x == 0) {
        float s = 0.0f;
        #pragma unroll
        for (int w = 0; w < 8; ++w) s += red[w];
        atomicAdd(&scal[0], s);       // n_sample
    }
}

// ============ K6: smoothed sizes -> new_mean ============
__global__ __launch_bounds__(128) void k_new_mean(
    const float* __restrict__ cluster_sum, const float* __restrict__ sums,
    const float* __restrict__ new_size, const float* __restrict__ scal,
    float* __restrict__ new_mean)
{
    const int k = blockIdx.x;      // 0..NCLUST-1
    const int d = threadIdx.x;
    const float nsamp = scal[0];
    const float smoothed = (new_size[k] + EPSV) * nsamp / (nsamp + (float)NCLUST * EPSV);
    const float nsum = cluster_sum[(size_t)k * DIM + d] * EMA
                     + sums[(size_t)k * DIM + d] * (1.0f - EMA);
    new_mean[(size_t)k * DIM + d] = nsum / smoothed;
}

// ============ K7: gather quantize, write outputs + idx, accumulate loss ============
__global__ __launch_bounds__(128) void k_gather_out(
    const float* __restrict__ samples, const int* __restrict__ idx,
    const float* __restrict__ new_mean, float* __restrict__ out, float* __restrict__ scal)
{
    const int i = blockIdx.x;      // sample
    const int d = threadIdx.x;     // dim
    const int k = idx[i];
    const float x = samples[(size_t)i * DIM + d];
    const float q = new_mean[(size_t)k * DIM + d];
    out[(size_t)i * DIM + d] = q;  // inputs + sg(q - inputs) == q
    float sq = (x - q) * (x - q);
    #pragma unroll
    for (int off = 16; off >= 1; off >>= 1) sq += __shfl_xor(sq, off, 32);
    __shared__ float red[4];
    if ((threadIdx.x & 31) == 0) red[threadIdx.x >> 5] = sq;
    __syncthreads();
    if (threadIdx.x == 0) {
        atomicAdd(&scal[1], red[0] + red[1] + red[2] + red[3]);
        out[OUT_IDX_OFF + i] = (float)k;
    }
}

// ============ K8: finalize loss ============
__global__ void k_loss(const float* __restrict__ scal, float* __restrict__ out)
{
    if (threadIdx.x == 0 && blockIdx.x == 0)
        out[OUT_LOSS_OFF] = BETAV * scal[1] / (float)(NSAMP * DIM);
}

extern "C" void kernel_launch(void* const* d_in, const int* in_sizes, int n_in,
                              void* d_out, int out_size, void* d_ws, size_t ws_size,
                              hipStream_t stream) {
    const float* inputs       = (const float*)d_in[0];   // [4,8,32,32,128]
    const float* cluster_mean = (const float*)d_in[1];   // [10000,128]
    const float* cluster_size = (const float*)d_in[2];   // [10000]
    const float* cluster_sum  = (const float*)d_in[3];   // [10000,128]
    float* out = (float*)d_out;

    char* ws = (char*)d_ws;
    _Float16* ef16   = (_Float16*)(ws + OFF_EF16);
    float*    en2    = (float*)(ws + OFF_EN2);
    int*      idx    = (int*)(ws + OFF_IDX);
    float*    counts = (float*)(ws + OFF_COUNTS);
    float*    sums   = (float*)(ws + OFF_SUMS);
    float*    nsize  = (float*)(ws + OFF_NSIZE);
    float*    nmean  = (float*)(ws + OFF_NMEAN);
    float*    scal   = (float*)(ws + OFF_SCAL);

    k_prep_clusters<<<NPAD, 128, 0, stream>>>(cluster_mean, ef16, en2);
    k_argmin_gemm<<<NSAMP / (8 * 16), 256, 0, stream>>>(inputs, ef16, en2, idx);
    k_zero<<<5000, 256, 0, stream>>>(counts, sums, scal);
    k_scatter<<<NSAMP, 128, 0, stream>>>(inputs, idx, counts, sums);
    k_ema_size<<<(NCLUST + 255) / 256, 256, 0, stream>>>(cluster_size, counts, nsize, scal);
    k_new_mean<<<NCLUST, 128, 0, stream>>>(cluster_sum, sums, nsize, scal, nmean);
    k_gather_out<<<NSAMP, 128, 0, stream>>>(inputs, idx, nmean, out, scal);
    k_loss<<<1, 32, 0, stream>>>(scal, out);
}
